// HybridGraphEncoder_35021163331784
// MI455X (gfx1250) — matrix operand order repaired and checked by reading, not compiled, not run on hardware
//
#include <hip/hip_runtime.h>
#include <hip/hip_bf16.h>
#include <math.h>

#define NN 30000
#define EE 600000
#define RR 32
#define NBASIS 30
#define DD 128
#define HH 8
#define NG 16

typedef __attribute__((ext_vector_type(16))) _Float16 v16h;
typedef __attribute__((ext_vector_type(8)))  _Float16 v8h;
typedef __attribute__((ext_vector_type(8)))  float    v8f;

// Fragment-major swizzle for the f16 B operand (K=row, N=col of a 128x128 matrix).
// Fragment f = (ct*4 + kk)*32 + lane holds 16 contiguous halves j=0..15 with
//   k = kk*32 + (lane>>4)*16 + j,  n = ct*16 + (lane&15)
// a contiguous k order, not the ISA table order; load_afrag uses the same k order for A.
__device__ __forceinline__ int bswz(int k, int n) {
  int kk = k >> 5, kl = k & 31;
  int g  = kl >> 4, j = kl & 15;
  int ct = n >> 4, rm = n & 15;
  return (((ct << 2) + kk) * 32 + (g * 16 + rm)) * 16 + j;
}

// ---------------- utility kernels ----------------
__global__ __launch_bounds__(256) void fill_f32_kernel(float* __restrict__ p, float v, int n) {
  int i = blockIdx.x * 256 + threadIdx.x;
  if (i < n) p[i] = v;
}

__global__ __launch_bounds__(256) void fill_u32_kernel(unsigned* __restrict__ p, unsigned v, int n) {
  int i = blockIdx.x * 256 + threadIdx.x;
  if (i < n) p[i] = v;
}

__global__ __launch_bounds__(256) void copy_f32_kernel(const float* __restrict__ s, float* __restrict__ d, int n) {
  int i = blockIdx.x * 256 + threadIdx.x;
  if (i < n) d[i] = s[i];
}

// x[n, :] = entity_emb[node_ids[n], :]
__global__ __launch_bounds__(256) void gather_x_kernel(const int* __restrict__ ids,
                                                       const float* __restrict__ emb,
                                                       float* __restrict__ x) {
  int i = blockIdx.x * 256 + threadIdx.x;      // N*32 float4 slots
  if (i >= NN * 32) return;
  int n = i >> 5;
  int c = (i & 31) * 4;
  float4 f = *(const float4*)(emb + (size_t)ids[n] * DD + c);
  *(float4*)(x + (size_t)n * DD + c) = f;
}

// ---------------- counting sort of edges by relation ----------------
__global__ __launch_bounds__(256) void hist_kernel(const int* __restrict__ et, int* __restrict__ hist) {
  int i = blockIdx.x * 256 + threadIdx.x;
  if (i < EE) atomicAdd(&hist[et[i]], 1);
}

__global__ void scan_kernel(const int* __restrict__ hist, int* __restrict__ off,
                            int* __restrict__ tileBase, int* __restrict__ cursor) {
  if (blockIdx.x != 0 || threadIdx.x != 0) return;
  int o = 0, tb = 0;
  for (int r = 0; r < RR; ++r) {
    off[r] = o; cursor[r] = o; tileBase[r] = tb;
    o  += hist[r];
    tb += (hist[r] + 15) >> 4;
  }
  off[RR] = o; tileBase[RR] = tb;
}

__global__ __launch_bounds__(256) void scatter_kernel(const int* __restrict__ et,
                                                      int* __restrict__ cursor,
                                                      int* __restrict__ perm) {
  int i = blockIdx.x * 256 + threadIdx.x;
  if (i >= EE) return;
  int pos = atomicAdd(&cursor[et[i]], 1);
  perm[pos] = i;
}

// edge counts per (tgt, rel) -> inverse (mean aggregation weights)
__global__ __launch_bounds__(256) void cnt_kernel(const int* __restrict__ ei, const int* __restrict__ et,
                                                  int* __restrict__ cnt) {
  int i = blockIdx.x * 256 + threadIdx.x;
  if (i >= EE) return;
  atomicAdd(&cnt[(size_t)ei[EE + i] * RR + et[i]], 1);
}

__global__ __launch_bounds__(256) void inv_kernel(int* __restrict__ ci, float* __restrict__ cf) {
  int i = blockIdx.x * 256 + threadIdx.x;
  if (i >= NN * RR) return;
  int c = ci[i];
  float f = (c > 0) ? (1.0f / (float)c) : 0.0f;
  cf[i] = f;
}

// ---------------- weight preparation (write fragment-swizzled f16) ----------------
// W[r] = sum_b comp[r,b] * bases[b]
__global__ __launch_bounds__(256) void rgcn_combine_kernel(const float* __restrict__ comp,
                                                           const float* __restrict__ bases,
                                                           _Float16* __restrict__ w16) {
  int i = blockIdx.x * 256 + threadIdx.x;
  if (i >= RR * DD * DD) return;
  int r  = i >> 14;
  int io = i & 16383;
  float s = 0.f;
#pragma unroll
  for (int b = 0; b < NBASIS; ++b) s += comp[r * NBASIS + b] * bases[(size_t)b * DD * DD + io];
  int k = io >> 7, n = io & 127;
  w16[(r << 14) + bswz(k, n)] = (_Float16)s;
}

__global__ __launch_bounds__(256) void convw_kernel(const float* __restrict__ W, _Float16* __restrict__ w16) {
  int i = blockIdx.x * 256 + threadIdx.x;
  if (i >= DD * DD) return;
  int k = i >> 7, n = i & 127;
  w16[bswz(k, n)] = (_Float16)W[i];
}

// ---------------- WMMA helpers ----------------
__device__ __forceinline__ v16h load_afrag(const _Float16* Alds, int rm, int g, int kk) {
  int base = rm * DD + kk * 32 + g * 16;
  v8h lo = *(const v8h*)(Alds + base);        // halves k .. k+7   (16B LDS load)
  v8h hi = *(const v8h*)(Alds + base + 8);    // halves k+8..k+15  (16B LDS load)
  v16h a;
#pragma unroll
  for (int j = 0; j < 8; ++j) { a[j] = lo[j]; a[8 + j] = hi[j]; }
  return a;
}

// ---------------- WMMA GEMM: C[M,128] = A[M,128] @ B16swz[128,128] + bias ----------------
// Requires M % 16 == 0 (true for all call sites: 30000, 32).
__global__ __launch_bounds__(256)
void gemm_rowblock_kernel(const float* __restrict__ A, const _Float16* __restrict__ B16,
                          const float* __restrict__ bias, float* __restrict__ C) {
  __shared__ _Float16 Alds[16 * DD];
  const int tid  = threadIdx.x;
  const int row0 = blockIdx.x * 16;
  // stage + convert A tile (16x128 f32 -> f16), full tile guaranteed valid
  for (int i = tid; i < 16 * 32; i += 256) {
    int r = i >> 5;
    int c = (i & 31) * 4;
    float4 f = *(const float4*)(A + (size_t)(row0 + r) * DD + c);
    Alds[r * DD + c + 0] = (_Float16)f.x;
    Alds[r * DD + c + 1] = (_Float16)f.y;
    Alds[r * DD + c + 2] = (_Float16)f.z;
    Alds[r * DD + c + 3] = (_Float16)f.w;
  }
  __syncthreads();

  const int wv   = tid >> 5;        // wave -> 16-col tile
  const int lane = tid & 31;
  const int g    = lane >> 4;
  const int rm   = lane & 15;
  const int col  = wv * 16 + rm;
  const float bsv = bias[col];      // hoisted, uniform per lane
  const v16h* Bf  = (const v16h*)B16;
  v8f acc = {};
#pragma unroll
  for (int kk = 0; kk < 4; ++kk) {
    v16h af = load_afrag(Alds, rm, g, kk);
    v16h bf = Bf[(wv * 4 + kk) * 32 + lane];   // 32B contiguous global load, L2-resident
    acc = __builtin_amdgcn_wmma_f32_16x16x32_f16(false, af, false, bf, (short)0, acc, false, false);
  }
  float* Cp = C + (size_t)(row0 + g * 8) * DD + col;
#pragma unroll
  for (int t = 0; t < 8; ++t) Cp[(size_t)t * DD] = acc[t] + bsv;   // coalesced across lanes
}

// ---------------- RGCN per-edge transform: out[tgt] += inv * (x[src] @ W[rel]) ----------------
__global__ __launch_bounds__(256)
void rgcn_edge_kernel(const float* __restrict__ x, const _Float16* __restrict__ Wall,
                      const int* __restrict__ ei, const int* __restrict__ perm,
                      const int* __restrict__ off, const int* __restrict__ tileBase,
                      const float* __restrict__ invc, float* __restrict__ out) {
  __shared__ _Float16 Alds[16 * DD];
  __shared__ int   tgt_s[16];
  __shared__ float sc_s[16];

  const int t = blockIdx.x;
  if (t >= tileBase[RR]) return;
  int r = 0;
  while (tileBase[r + 1] <= t) ++r;           // 32-entry scan, scalar-cached
  const int e0   = off[r] + (t - tileBase[r]) * 16;
  const int eEnd = off[r + 1];
  const int tid  = threadIdx.x;

  if (tid < 16) {
    int idx = e0 + tid;
    int tg = -1; float sc = 0.f;
    if (idx < eEnd) {
      int e = perm[idx];
      tg = ei[EE + e];
      sc = invc[(size_t)tg * RR + r];
    }
    tgt_s[tid] = tg;
    sc_s[tid]  = sc;
  }
  for (int i = tid; i < 16 * 32; i += 256) {   // gather 16 src rows, f32->f16
    int rr = i >> 5;
    int c  = (i & 31) * 4;
    float4 f = make_float4(0.f, 0.f, 0.f, 0.f);
    int idx = e0 + rr;
    if (idx < eEnd) {
      int e = perm[idx];
      int s = ei[e];
      f = *(const float4*)(x + (size_t)s * DD + c);
    }
    Alds[rr * DD + c + 0] = (_Float16)f.x;
    Alds[rr * DD + c + 1] = (_Float16)f.y;
    Alds[rr * DD + c + 2] = (_Float16)f.z;
    Alds[rr * DD + c + 3] = (_Float16)f.w;
  }
  __syncthreads();

  const int wv   = tid >> 5;
  const int lane = tid & 31;
  const int g    = lane >> 4;
  const int rm   = lane & 15;
  const int col  = wv * 16 + rm;
  const v16h* Bf = (const v16h*)(Wall + ((size_t)r << 14));
  v8f acc = {};
#pragma unroll
  for (int kk = 0; kk < 4; ++kk) {
    v16h af = load_afrag(Alds, rm, g, kk);
    v16h bf = Bf[(wv * 4 + kk) * 32 + lane];
    acc = __builtin_amdgcn_wmma_f32_16x16x32_f16(false, af, false, bf, (short)0, acc, false, false);
  }
#pragma unroll
  for (int tt = 0; tt < 8; ++tt) {
    int m  = g * 8 + tt;
    int tg = tgt_s[m];
    if (tg >= 0) atomicAdd(&out[(size_t)tg * DD + col], acc[tt] * sc_s[m]);
  }
}

// ---------------- attention (edge-parallel VALU) ----------------
__device__ inline void atomicMaxFloat(float* addr, float val) {
  if (val >= 0.f) atomicMax((int*)addr, __float_as_int(val));
  else            atomicMin((unsigned int*)addr, __float_as_uint(val));
}

__global__ __launch_bounds__(256)
void attn_alpha_kernel(const float* __restrict__ q, const float* __restrict__ k,
                       const float* __restrict__ erel, const int* __restrict__ ei,
                       const int* __restrict__ et, float* __restrict__ alpha,
                       float* __restrict__ mbuf) {
  int id = blockIdx.x * 256 + threadIdx.x;    // E*H
  if (id >= EE * HH) return;
  int e = id >> 3, h = id & 7;
  int s = ei[e], tg = ei[EE + e], r = et[e];
  const float4* qp = (const float4*)(q + (size_t)tg * DD + h * 16);
  const float4* kp = (const float4*)(k + (size_t)s  * DD + h * 16);
  const float4* ep = (const float4*)(erel + (size_t)r * DD + h * 16);
  float acc = 0.f;
#pragma unroll
  for (int i = 0; i < 4; ++i) {
    float4 qq = qp[i], kk = kp[i], ee = ep[i];
    acc += qq.x * (kk.x + ee.x) + qq.y * (kk.y + ee.y) + qq.z * (kk.z + ee.z) + qq.w * (kk.w + ee.w);
  }
  acc *= 0.25f;                                // 1/sqrt(C), C=16
  alpha[id] = acc;
  atomicMaxFloat(&mbuf[(size_t)tg * HH + h], acc);
}

__global__ __launch_bounds__(256)
void attn_expsum_kernel(const int* __restrict__ ei, float* __restrict__ alpha,
                        const float* __restrict__ mbuf, float* __restrict__ denom) {
  int id = blockIdx.x * 256 + threadIdx.x;
  if (id >= EE * HH) return;
  int e = id >> 3, h = id & 7;
  int tg = ei[EE + e];
  float a = __expf(alpha[id] - mbuf[(size_t)tg * HH + h]);
  alpha[id] = a;
  atomicAdd(&denom[(size_t)tg * HH + h], a);
}

__global__ __launch_bounds__(256)
void attn_agg_kernel(const float* __restrict__ v, const float* __restrict__ erel,
                     const int* __restrict__ ei, const int* __restrict__ et,
                     const float* __restrict__ alpha, const float* __restrict__ denom,
                     float* __restrict__ out) {
  int id = blockIdx.x * 256 + threadIdx.x;    // E*32 (float4 per thread)
  if (id >= EE * 32) return;
  int e = id >> 5;
  int d0 = (id & 31) * 4;
  int h = d0 >> 4;
  int s = ei[e], tg = ei[EE + e], r = et[e];
  float w = alpha[(size_t)e * HH + h] / fmaxf(denom[(size_t)tg * HH + h], 1e-16f);
  float4 vv = *(const float4*)(v + (size_t)s * DD + d0);
  float4 ee = *(const float4*)(erel + (size_t)r * DD + d0);
  float* o = out + (size_t)tg * DD + d0;
  atomicAdd(o + 0, (vv.x + ee.x) * w);
  atomicAdd(o + 1, (vv.y + ee.y) * w);
  atomicAdd(o + 2, (vv.z + ee.z) * w);
  atomicAdd(o + 3, (vv.w + ee.w) * w);
}

// ---------------- layernorm + relu (+ residual) ----------------
__global__ __launch_bounds__(128)
void ln_relu_kernel(const float* __restrict__ in, float* __restrict__ xbuf,
                    const float* __restrict__ g, const float* __restrict__ b, int residual) {
  __shared__ float red[128];
  int n = blockIdx.x;
  int t = threadIdx.x;
  float v = in[(size_t)n * DD + t];
  red[t] = v; __syncthreads();
  for (int s = 64; s > 0; s >>= 1) { if (t < s) red[t] += red[t + s]; __syncthreads(); }
  float mu = red[0] / 128.f; __syncthreads();
  float dv = v - mu;
  red[t] = dv * dv; __syncthreads();
  for (int s = 64; s > 0; s >>= 1) { if (t < s) red[t] += red[t + s]; __syncthreads(); }
  float var = red[0] / 128.f;
  float y = dv * rsqrtf(var + 1e-5f) * g[t] + b[t];
  y = fmaxf(y, 0.f);
  if (residual) xbuf[(size_t)n * DD + t] += y;
  else          xbuf[(size_t)n * DD + t]  = y;
}

// ---------------- graph pooling ----------------
__global__ __launch_bounds__(256) void pool_acc_kernel(const float* __restrict__ x,
                                                       const int* __restrict__ batch,
                                                       float* __restrict__ gacc) {
  int i = blockIdx.x * 256 + threadIdx.x;    // N*128
  if (i >= NN * DD) return;
  int n = i >> 7, d = i & 127;
  atomicAdd(&gacc[(size_t)batch[n] * DD + d], x[i]);
}

__global__ __launch_bounds__(256) void pool_cnt_kernel(const int* __restrict__ batch, float* __restrict__ gcnt) {
  int i = blockIdx.x * 256 + threadIdx.x;
  if (i < NN) atomicAdd(&gcnt[batch[i]], 1.f);
}

__global__ __launch_bounds__(256) void pool_out_kernel(const float* __restrict__ gacc,
                                                       const float* __restrict__ gcnt,
                                                       float* __restrict__ out) {
  int i = blockIdx.x * 256 + threadIdx.x;    // NG*128
  if (i >= NG * DD) return;
  int gidx = i >> 7;
  out[i] = gacc[i] / fmaxf(gcnt[gidx], 1.f);
}

// ---------------- launch ----------------
extern "C" void kernel_launch(void* const* d_in, const int* in_sizes, int n_in,
                              void* d_out, int out_size, void* d_ws, size_t ws_size,
                              hipStream_t stream) {
  (void)in_sizes; (void)n_in; (void)out_size; (void)ws_size;
  const int*   node_ids = (const int*)d_in[0];
  const int*   ei       = (const int*)d_in[1];      // [2,E] flat: src=ei[e], tgt=ei[E+e]
  const int*   et       = (const int*)d_in[2];
  const int*   batch    = (const int*)d_in[3];
  const float* ent_emb  = (const float*)d_in[4];
  const float* rel_emb  = (const float*)d_in[5];
  const float* comp     = (const float*)d_in[6];
  const float* bases    = (const float*)d_in[7];
  const float* root     = (const float*)d_in[8];
  const float* rbias    = (const float*)d_in[9];
  const float* rlng     = (const float*)d_in[10];
  const float* rlnb     = (const float*)d_in[11];
  const float* wq = (const float*)d_in[12]; const float* bq = (const float*)d_in[13];
  const float* wk = (const float*)d_in[14]; const float* bk = (const float*)d_in[15];
  const float* wv = (const float*)d_in[16]; const float* bv = (const float*)d_in[17];
  const float* we = (const float*)d_in[18];
  const float* ws_ = (const float*)d_in[19]; const float* bs_ = (const float*)d_in[20];
  const float* tlng = (const float*)d_in[21]; const float* tlnb = (const float*)d_in[22];
  float* out = (float*)d_out;

  // ---- workspace partition (256B-aligned sub-allocations) ----
  size_t off_ = 0;
  auto alloc = [&](size_t bytes) -> void* {
    off_ = (off_ + 255) & ~(size_t)255;
    void* p = (char*)d_ws + off_;
    off_ += bytes;
    return p;
  };
  float*     x     = (float*)alloc((size_t)NN * DD * 4);
  float*     tmp   = (float*)alloc((size_t)NN * DD * 4);
  float*     qb    = (float*)alloc((size_t)NN * DD * 4);
  float*     kb    = (float*)alloc((size_t)NN * DD * 4);
  float*     vb    = (float*)alloc((size_t)NN * DD * 4);
  float*     invb  = (float*)alloc((size_t)NN * RR * 4);   // int counts then float inverse, in place
  int*       perm  = (int*)  alloc((size_t)EE * 4);
  float*     alpha = (float*)alloc((size_t)EE * HH * 4);
  float*     mbuf  = (float*)alloc((size_t)NN * HH * 4);
  float*     denom = (float*)alloc((size_t)NN * HH * 4);
  _Float16*  W16r  = (_Float16*)alloc((size_t)RR * DD * DD * 2);
  _Float16*  Wn16  = (_Float16*)alloc((size_t)DD * DD * 2);
  float*     erel  = (float*)alloc((size_t)RR * DD * 4);
  float*     zbias = (float*)alloc(DD * 4);
  int*       hist  = (int*)alloc(32 * 4);
  int*       eoff  = (int*)alloc(33 * 4);
  int*       tileB = (int*)alloc(33 * 4);
  int*       curs  = (int*)alloc(32 * 4);
  float*     gacc  = (float*)alloc((size_t)NG * DD * 4);
  float*     gcnt  = (float*)alloc(NG * 4);

  const int gN16   = NN / 16;                        // 1875 row-blocks (exact)
  const int gEdgeT = (EE + 15) / 16 + RR;            // upper bound on relation tiles
  const int gE     = (EE + 255) / 256;
  const int gEH    = (EE * HH) / 256;                // 18750
  const int gE32   = (EE * 32) / 256;                // 75000
  const int gND    = (NN * DD) / 256;                // 15000
  const int gNR    = (NN * RR + 255) / 256;
  const int gNH    = (NN * HH + 255) / 256;

  // ---- static graph preprocessing (recomputed every call; deterministic work) ----
  fill_u32_kernel<<<1, 256, 0, stream>>>((unsigned*)hist, 0u, 32);
  fill_f32_kernel<<<1, 256, 0, stream>>>(zbias, 0.f, DD);
  hist_kernel<<<gE, 256, 0, stream>>>(et, hist);
  scan_kernel<<<1, 1, 0, stream>>>(hist, eoff, tileB, curs);
  scatter_kernel<<<gE, 256, 0, stream>>>(et, curs, perm);
  fill_u32_kernel<<<gNR, 256, 0, stream>>>((unsigned*)invb, 0u, NN * RR);
  cnt_kernel<<<gE, 256, 0, stream>>>(ei, et, (int*)invb);
  inv_kernel<<<gNR, 256, 0, stream>>>((int*)invb, invb);
  gather_x_kernel<<<(NN * 32 + 255) / 256, 256, 0, stream>>>(node_ids, ent_emb, x);

  // ---- RGCN layers ----
  for (int l = 0; l < 2; ++l) {
    rgcn_combine_kernel<<<(RR * DD * DD) / 256, 256, 0, stream>>>(
        comp + (size_t)l * RR * NBASIS, bases + (size_t)l * NBASIS * DD * DD, W16r);
    convw_kernel<<<64, 256, 0, stream>>>(root + (size_t)l * DD * DD, Wn16);
    gemm_rowblock_kernel<<<gN16, 256, 0, stream>>>(x, Wn16, rbias + l * DD, tmp);
    rgcn_edge_kernel<<<gEdgeT, 256, 0, stream>>>(x, W16r, ei, perm, eoff, tileB, invb, tmp);
    ln_relu_kernel<<<NN, 128, 0, stream>>>(tmp, x, rlng + l * DD, rlnb + l * DD, 0);
  }

  // ---- TransformerConv layers ----
  for (int l = 0; l < 2; ++l) {
    convw_kernel<<<64, 256, 0, stream>>>(wq + (size_t)l * DD * DD, Wn16);
    gemm_rowblock_kernel<<<gN16, 256, 0, stream>>>(x, Wn16, bq + l * DD, qb);
    convw_kernel<<<64, 256, 0, stream>>>(wk + (size_t)l * DD * DD, Wn16);
    gemm_rowblock_kernel<<<gN16, 256, 0, stream>>>(x, Wn16, bk + l * DD, kb);
    convw_kernel<<<64, 256, 0, stream>>>(wv + (size_t)l * DD * DD, Wn16);
    gemm_rowblock_kernel<<<gN16, 256, 0, stream>>>(x, Wn16, bv + l * DD, vb);
    convw_kernel<<<64, 256, 0, stream>>>(we + (size_t)l * DD * DD, Wn16);
    gemm_rowblock_kernel<<<RR / 16, 256, 0, stream>>>(rel_emb, Wn16, zbias, erel);
    convw_kernel<<<64, 256, 0, stream>>>(ws_ + (size_t)l * DD * DD, Wn16);
    gemm_rowblock_kernel<<<gN16, 256, 0, stream>>>(x, Wn16, bs_ + l * DD, tmp);

    fill_f32_kernel<<<gNH, 256, 0, stream>>>(mbuf, -INFINITY, NN * HH);
    fill_f32_kernel<<<gNH, 256, 0, stream>>>(denom, 0.f, NN * HH);
    attn_alpha_kernel<<<gEH, 256, 0, stream>>>(qb, kb, erel, ei, et, alpha, mbuf);
    attn_expsum_kernel<<<gEH, 256, 0, stream>>>(ei, alpha, mbuf, denom);
    attn_agg_kernel<<<gE32, 256, 0, stream>>>(vb, erel, ei, et, alpha, denom, tmp);
    ln_relu_kernel<<<NN, 128, 0, stream>>>(tmp, x, tlng + l * DD, tlnb + l * DD, 1);
  }

  // ---- outputs: x then graph mean-pool ----
  copy_f32_kernel<<<gND, 256, 0, stream>>>(x, out, NN * DD);
  fill_f32_kernel<<<(NG * DD + 255) / 256, 256, 0, stream>>>(gacc, 0.f, NG * DD);
  fill_f32_kernel<<<1, 256, 0, stream>>>(gcnt, 0.f, NG);
  pool_acc_kernel<<<gND, 256, 0, stream>>>(x, batch, gacc);
  pool_cnt_kernel<<<(NN + 255) / 256, 256, 0, stream>>>(batch, gcnt);
  pool_out_kernel<<<(NG * DD + 255) / 256, 256, 0, stream>>>(gacc, gcnt, out + (size_t)NN * DD);
}
